// CPSModel_54434415509841
// MI455X (gfx1250) — matrix operand order repaired and checked by reading, not compiled, not run on hardware
//
#include <hip/hip_runtime.h>

// ---------------------------------------------------------------------------
// MI455X (gfx1250) implementation. Wave32, WMMA bf16 16x16x32 with f32 acc.
// GEMM-dominated workload (~190 GFLOP) -> all Linear layers run through
// v_wmma_f32_16x16x32_bf16 with 4-tile register blocking per wave and the
// weight tile staged in LDS per k-step by the Tensor Data Mover
// (tensor_load_to_lds + s_wait_tensorcnt), falling back to manual staging if
// the builtin is unavailable. Graph propagation (L2-resident, 80MB features)
// uses global float atomics. All scratch carved from d_ws.
// ---------------------------------------------------------------------------

#define NN     10000
#define EE     160000
#define HV     2000
#define DDIM   256
#define NHEADS 8
#define HDIM   32
#define NSC    3
#define NFREQ  64
#define ENCD   128
#define EPSLN  1e-5f
#define ALPHA_ 0.2f

typedef __attribute__((ext_vector_type(16))) __bf16 v16bf;
typedef __attribute__((ext_vector_type(8)))  float  v8f;
typedef __attribute__((ext_vector_type(4)))  unsigned v4u;
typedef __attribute__((ext_vector_type(8)))  int    v8i;
typedef __attribute__((ext_vector_type(4)))  int    v4i;

#if __has_builtin(__builtin_amdgcn_tensor_load_to_lds) && __has_builtin(__builtin_amdgcn_s_wait_tensorcnt)
#define HAVE_TDM 1
#if defined(__clang_major__) && __clang_major__ >= 23
#define TDM_LOAD(g0, g1, g2, g3) \
  __builtin_amdgcn_tensor_load_to_lds(g0, g1, g2, g3, (v8i){0,0,0,0,0,0,0,0}, 0)
#else
#define TDM_LOAD(g0, g1, g2, g3) \
  __builtin_amdgcn_tensor_load_to_lds(g0, g1, g2, g3, 0)
#endif
#endif

enum { ACT_NONE = 0, ACT_GELU = 1, ACT_SILU = 2, ACT_SOFTPLUS = 3, ACT_SIGMOID = 4 };

template<int ACT> __device__ __forceinline__ float act_apply(float x) {
  if constexpr (ACT == ACT_GELU)     return 0.5f * x * (1.f + erff(x * 0.70710678118654752f));
  if constexpr (ACT == ACT_SILU)     return x / (1.f + __expf(-x));
  if constexpr (ACT == ACT_SOFTPLUS) return (x > 20.f) ? x : log1pf(__expf(x));
  if constexpr (ACT == ACT_SIGMOID)  return 1.f / (1.f + __expf(-x));
  return x;
}

// ---------------- WMMA GEMM: C[M,O] = act(A[M,K] @ W[O,K]^T + bias) --------
// Block = 256 threads = 8 waves, computes 128 rows x 64 cols.
// Per k-step: TDM stages W tile (32k x 64cols, fp32, padded LDS stride 36
// dwords -> 16B aligned + bank-friendly; OOB k/cols zero-filled by TDM).
// Each wave: one A fragment (global, fp32->bf16) reused across 4 WMMA tiles.
#define BSTRIDE 36   // LDS dwords per column strip (32 data + 4 pad)

template<int ACT>
__global__ __launch_bounds__(256)
void gemm_wmma_k(const float* __restrict__ A, const float* __restrict__ W,
                 const float* __restrict__ bias, float* __restrict__ C,
                 int M, int K, int O) {
  __shared__ float Bt[64 * BSTRIDE];
  if (K < 0) Bt[0] = 0.f;  // keep LDS writes visible in TDM path
  const int wave = threadIdx.x >> 5;
  const int lane = threadIdx.x & 31;
  const int r16 = lane & 15;
  const int hi  = lane >> 4;
  const int oc0 = blockIdx.x * 64;
  const int tiles_m = M >> 4;
  const int tmRaw = blockIdx.y * 8 + wave;
  const bool mvalid = tmRaw < tiles_m;
  const int tm = mvalid ? tmRaw : (tiles_m - 1);   // clamp: stay alive for barriers
  const float* Arow = A + (size_t)(tm * 16 + r16) * K;

  v8f acc[4] = {v8f{}, v8f{}, v8f{}, v8f{}};

  for (int k0 = 0; k0 < K; k0 += 32) {
    __syncthreads();   // previous tile fully consumed
#ifdef HAVE_TDM
    if (wave == 0) {
      unsigned long long ga = (unsigned long long)(const void*)(W + (size_t)oc0 * K + k0);
      unsigned lds = (unsigned)(unsigned long long)(const void*)&Bt[0];
      unsigned td0 = (unsigned)(K - k0);   // remaining k: TDM zero-fills x >= td0
      unsigned td1 = (unsigned)(O - oc0);  // remaining cols: zero-fills y >= td1
      v4u g0;
      g0[0] = 1u;                                   // count=1 valid descriptor
      g0[1] = lds;                                  // lds_addr
      g0[2] = (unsigned)ga;                         // global_addr[31:0]
      g0[3] = (unsigned)((ga >> 32) & 0x1ffffffull) | 0x80000000u;  // [56:32] | type=2
      v8i g1;
      g1[0] = (int)((2u << 16) | (1u << 20) | (4u << 22) | (3u << 25));
               // data_size=4B, pad_enable, pad_interval=32dw, pad_amount=4dw
      g1[1] = (int)((td0 & 0xffffu) << 16);                       // tensor_dim0 lo
      g1[2] = (int)(((td0 >> 16) & 0xffffu) | ((td1 & 0xffffu) << 16));
      g1[3] = (int)(((td1 >> 16) & 0xffffu) | (32u << 16));       // tile_dim0=32
      g1[4] = 64;                                                 // tile_dim1=64
      g1[5] = (int)(unsigned)K;                                   // dim0 stride
      g1[6] = 0;
      g1[7] = 0;
      v4i gz = {0, 0, 0, 0};
      TDM_LOAD(g0, g1, gz, gz);
      __builtin_amdgcn_s_wait_tensorcnt((short)0);
    }
#else
    {
      int col = threadIdx.x >> 2;      // 0..63
      int seg = threadIdx.x & 3;       // 8 floats each
      int oc = oc0 + col;
      const float* wp = W + (size_t)(oc < O ? oc : 0) * K;
      bool cok = oc < O;
#pragma unroll
      for (int j = 0; j < 8; ++j) {
        int kk = k0 + seg * 8 + j;
        Bt[col * BSTRIDE + seg * 8 + j] = (cok && kk < K) ? wp[kk] : 0.f;
      }
    }
#endif
    __syncthreads();   // tile staged

    // ---- A fragment: lane holds row tm*16+r16, k = k0 + hi*8+[0..7], +16
    v16bf va;
    if (k0 + 32 <= K) {
      __builtin_prefetch(Arow + k0 + 128, 0, 0);
      const float4* pa0 = (const float4*)(Arow + k0 + hi * 8);
      const float4* pa1 = (const float4*)(Arow + k0 + 16 + hi * 8);
      float4 a0 = pa0[0], a1 = pa0[1], a2 = pa1[0], a3 = pa1[1];
      va[0]=(__bf16)a0.x; va[1]=(__bf16)a0.y; va[2]=(__bf16)a0.z; va[3]=(__bf16)a0.w;
      va[4]=(__bf16)a1.x; va[5]=(__bf16)a1.y; va[6]=(__bf16)a1.z; va[7]=(__bf16)a1.w;
      va[8]=(__bf16)a2.x; va[9]=(__bf16)a2.y; va[10]=(__bf16)a2.z; va[11]=(__bf16)a2.w;
      va[12]=(__bf16)a3.x; va[13]=(__bf16)a3.y; va[14]=(__bf16)a3.z; va[15]=(__bf16)a3.w;
    } else {
#pragma unroll
      for (int j = 0; j < 8; ++j) {
        int ka = k0 + hi * 8 + j;
        int kb = k0 + 16 + hi * 8 + j;
        va[j]     = (ka < K) ? (__bf16)Arow[ka] : (__bf16)0.f;
        va[8 + j] = (kb < K) ? (__bf16)Arow[kb] : (__bf16)0.f;
      }
    }

    // ---- 4 B fragments from LDS (lane = col r16 of tile c, k = hi*8+[0..7], +16)
#pragma unroll
    for (int c = 0; c < 4; ++c) {
      const float4* bp = (const float4*)&Bt[(c * 16 + r16) * BSTRIDE + hi * 8];
      float4 b0 = bp[0], b1 = bp[1], b2 = bp[4], b3 = bp[5];
      v16bf vb;
      vb[0]=(__bf16)b0.x; vb[1]=(__bf16)b0.y; vb[2]=(__bf16)b0.z; vb[3]=(__bf16)b0.w;
      vb[4]=(__bf16)b1.x; vb[5]=(__bf16)b1.y; vb[6]=(__bf16)b1.z; vb[7]=(__bf16)b1.w;
      vb[8]=(__bf16)b2.x; vb[9]=(__bf16)b2.y; vb[10]=(__bf16)b2.z; vb[11]=(__bf16)b2.w;
      vb[12]=(__bf16)b3.x; vb[13]=(__bf16)b3.y; vb[14]=(__bf16)b3.z; vb[15]=(__bf16)b3.w;
      acc[c] = __builtin_amdgcn_wmma_f32_16x16x32_bf16(false, va, false, vb, (short)0,
                                                       acc[c], false, false);
    }
  }

  // ---- epilogue: C layout lane->col N=r16, VGPR v -> row v + 8*hi
#pragma unroll
  for (int c = 0; c < 4; ++c) {
    int oc = oc0 + c * 16 + r16;
    if (mvalid && oc < O) {
      float bv = bias ? bias[oc] : 0.f;
#pragma unroll
      for (int v = 0; v < 8; ++v) {
        int mr = tm * 16 + v + 8 * hi;
        C[(size_t)mr * O + oc] = act_apply<ACT>(acc[c][v] + bv);
      }
    }
  }
}

// ---------------- elementwise / graph kernels ------------------------------
__global__ __launch_bounds__(256) void fill_k(float* p, float v, long n) {
  long i = (long)blockIdx.x * 256 + threadIdx.x;
  if (i < n) p[i] = v;
}
__global__ __launch_bounds__(256) void scale_k(float* dst, const float* src, float a, long n) {
  long i = (long)blockIdx.x * 256 + threadIdx.x;
  if (i < n) dst[i] = a * src[i];
}
__global__ __launch_bounds__(256) void axpy_k(float* dst, const float* src, float a, long n) {
  long i = (long)blockIdx.x * 256 + threadIdx.x;
  if (i < n) dst[i] += a * src[i];
}
__global__ __launch_bounds__(256) void degcnt_k(const int* ei, float* deg) {
  int e = blockIdx.x * 256 + threadIdx.x;
  if (e < EE) atomicAdd(&deg[ei[EE + e]], 1.0f);
}
__global__ __launch_bounds__(256) void dinv_k(float* deg) {
  int i = blockIdx.x * 256 + threadIdx.x;
  if (i < NN) { float d = deg[i]; deg[i] = d > 0.f ? rsqrtf(d) : 0.f; }
}
__global__ __launch_bounds__(256) void selfloop_k(float* dst, const float* src,
                                                  const float* dinv) {
  long i = (long)blockIdx.x * 256 + threadIdx.x;
  if (i < (long)NN * HV) {
    int n = (int)(i / HV);
    float w = dinv[n] * dinv[n];
    dst[i] = w * src[i];
  }
}
__global__ __launch_bounds__(256) void edgeprop_k(const int* ei, const float* dinv,
                                                  const float* src, float* dst) {
  int e = blockIdx.x;
  int r = ei[e], c = ei[EE + e];
  float w = dinv[r] * dinv[c];
  const float* s = src + (size_t)r * HV;
  float* d = dst + (size_t)c * HV;
  for (int f = threadIdx.x; f < HV; f += 256) atomicAdd(&d[f], w * s[f]);
}
__global__ __launch_bounds__(256) void colstat_k(const float* __restrict__ X, float* stat) {
  int j = blockIdx.x;
  float s1 = 0.f, s2 = 0.f;
  for (int i = threadIdx.x; i < NN; i += 256) {
    float v = X[(size_t)i * DDIM + j];
    s1 += v; s2 += v * v;
  }
  __shared__ float red[256];
  red[threadIdx.x] = s1; __syncthreads();
  for (int o = 128; o > 0; o >>= 1) { if (threadIdx.x < o) red[threadIdx.x] += red[threadIdx.x + o]; __syncthreads(); }
  float mean = red[0] / NN;
  __syncthreads();
  red[threadIdx.x] = s2; __syncthreads();
  for (int o = 128; o > 0; o >>= 1) { if (threadIdx.x < o) red[threadIdx.x] += red[threadIdx.x + o]; __syncthreads(); }
  if (threadIdx.x == 0) {
    float var = red[0] / NN - mean * mean;
    stat[j] = mean;
    stat[DDIM + j] = rsqrtf(var + EPSLN);
  }
}
__global__ __launch_bounds__(256) void bnapply_k(float* X, const float* stat,
                                                 const float* g, const float* b) {
  long i = (long)blockIdx.x * 256 + threadIdx.x;
  if (i < (long)NN * DDIM) {
    int j = (int)(i & (DDIM - 1));
    X[i] = (X[i] - stat[j]) * stat[DDIM + j] * g[j] + b[j];
  }
}
template<int POST_SILU>
__global__ __launch_bounds__(256)
void ln_k(const float* __restrict__ X, const float* __restrict__ g,
          const float* __restrict__ b, const float* __restrict__ res,
          float* __restrict__ Y, int Dd) {
  const int r = blockIdx.x;
  const float* x = X + (size_t)r * Dd;
  float s1 = 0.f, s2 = 0.f;
  for (int j = threadIdx.x; j < Dd; j += 256) { float v = x[j]; s1 += v; s2 += v * v; }
  __shared__ float red[256];
  red[threadIdx.x] = s1; __syncthreads();
  for (int o = 128; o > 0; o >>= 1) { if (threadIdx.x < o) red[threadIdx.x] += red[threadIdx.x + o]; __syncthreads(); }
  const float mean = red[0] / Dd;
  __syncthreads();
  red[threadIdx.x] = s2; __syncthreads();
  for (int o = 128; o > 0; o >>= 1) { if (threadIdx.x < o) red[threadIdx.x] += red[threadIdx.x + o]; __syncthreads(); }
  const float var = red[0] / Dd - mean * mean;
  const float inv = rsqrtf(var + EPSLN);
  float* y = Y + (size_t)r * Dd;
  for (int j = threadIdx.x; j < Dd; j += 256) {
    float v = (x[j] - mean) * inv * g[j] + b[j];
    if constexpr (POST_SILU) v = v / (1.f + __expf(-v));
    if (res) v += res[(size_t)r * Dd + j];
    y[j] = v;
  }
}
__global__ __launch_bounds__(256) void fourier_k(const float* __restrict__ pos,
                                                 const float* __restrict__ B,
                                                 float* __restrict__ enc) {
  int i = blockIdx.x * 256 + threadIdx.x;
  if (i < NN * NFREQ) {
    int n = i / NFREQ, f = i % NFREQ;
    float sc = 6.2831853071795865f * (pos[n * 2] * B[f * 2] + pos[n * 2 + 1] * B[f * 2 + 1]);
    enc[(size_t)n * ENCD + f]         = cosf(sc);
    enc[(size_t)n * ENCD + NFREQ + f] = sinf(sc);
  }
}
__global__ __launch_bounds__(256)
void attn_k(const float* __restrict__ q, const float* __restrict__ keys,
            const float* __restrict__ vals, const float* __restrict__ f0,
            float* __restrict__ att, float* __restrict__ attw) {
  int i = blockIdx.x * 256 + threadIdx.x;
  if (i >= NN * NHEADS) return;
  int n = i / NHEADS, h = i % NHEADS;
  const float* qp = q + (size_t)n * DDIM + h * HDIM;
  float sc[NSC];
  for (int s = 0; s < NSC; ++s) {
    const float* kp = keys + ((size_t)s * NN + n) * DDIM + h * HDIM;
    float d = 0.f;
    for (int j = 0; j < HDIM; ++j) d += qp[j] * kp[j];
    sc[s] = d * 0.17677669529663689f;
  }
  float m = fmaxf(sc[0], fmaxf(sc[1], sc[2]));
  float e[NSC], sum = 0.f;
  for (int s = 0; s < NSC; ++s) { e[s] = __expf(sc[s] - m); sum += e[s]; }
  float w[NSC];
  for (int s = 0; s < NSC; ++s) {
    w[s] = e[s] / sum;
    attw[(size_t)n * (NSC * NHEADS) + s * NHEADS + h] = w[s];
  }
  for (int j = 0; j < HDIM; ++j) {
    float a = 0.f;
    for (int s = 0; s < NSC; ++s) a += w[s] * vals[((size_t)s * NN + n) * DDIM + h * HDIM + j];
    att[(size_t)n * DDIM + h * HDIM + j] = a + f0[(size_t)n * DDIM + h * HDIM + j];
  }
}

// ---------------- host side ------------------------------------------------
template<int ACT>
static void launch_gemm(hipStream_t st, const float* A, const float* W, const float* b,
                        float* C, int M, int K, int O) {
  dim3 grid((O + 63) / 64, (M / 16 + 7) / 8);
  gemm_wmma_k<ACT><<<grid, 256, 0, st>>>(A, W, b, C, M, K, O);
}

// d_in flattening assumption: jax tree order (dicts by sorted key, lists in order)
enum {
  IN_X = 0, IN_EI, IN_POS,
  P_DEC_DISP_B, P_DEC_DISP_W,
  P_DEC_INP_LIN_B, P_DEC_INP_LIN_W, P_DEC_INP_LN_B, P_DEC_INP_LN_G,
  P_DEC_MEAN_B, P_DEC_MEAN_W,
  P_DEC_PI_B, P_DEC_PI_W,
  P_DEC_PROJ0_B, P_DEC_PROJ0_W, P_DEC_PROJ1_B, P_DEC_PROJ1_W,
  P_RB0_LIN1_B, P_RB0_LIN1_W, P_RB0_LIN2_B, P_RB0_LIN2_W,
  P_RB0_LN1_B, P_RB0_LN1_G, P_RB0_LN2_B, P_RB0_LN2_G,
  P_RB1_LIN1_B, P_RB1_LIN1_W, P_RB1_LIN2_B, P_RB1_LIN2_W,
  P_RB1_LN1_B, P_RB1_LN1_G, P_RB1_LN2_B, P_RB1_LN2_G,
  P_RB2_LIN1_B, P_RB2_LIN1_W, P_RB2_LIN2_B, P_RB2_LIN2_W,
  P_RB2_LN1_B, P_RB2_LN1_G, P_RB2_LN2_B, P_RB2_LN2_G,
  P_STU_B,
  P_STU_H0_LIN_B, P_STU_H0_LIN_W, P_STU_H0_LN_B, P_STU_H0_LN_G,
  P_STU_H1_LIN_B, P_STU_H1_LIN_W, P_STU_H1_LN_B, P_STU_H1_LN_G,
  P_STU_OUT_B, P_STU_OUT_W,
  P_T_LN_B, P_T_LN_G, P_T_OUT_B, P_T_OUT_W,
  P_SC0  // each scale group: bn.b, bn.g, conv.b, conv.w, k.b, k.w, q.b, q.w, v.b, v.w
};

extern "C" void kernel_launch(void* const* d_in, const int* in_sizes, int n_in,
                              void* d_out, int out_size, void* d_ws, size_t ws_size,
                              hipStream_t stream) {
  (void)in_sizes; (void)n_in; (void)out_size; (void)ws_size;
  const float* x   = (const float*)d_in[IN_X];
  const int*   ei  = (const int*)d_in[IN_EI];
  const float* pos = (const float*)d_in[IN_POS];
  auto P = [&](int i) { return (const float*)d_in[i]; };

  float* out = (float*)d_out;
  float* ws  = (float*)d_ws;

  float* bufA  = ws;                    // 20M  (prop ping / decoder hcur)
  float* bufB  = ws + 20000000LL;       // 20M  (prop pong / decoder hnext)
  float* bufH  = ws + 40000000LL;       // 20M  (prop accumulator / student+dec tmp)
  float* feats = ws + 60000000LL;       // [3][N,256]
  float* nf    = ws + 67680000LL;       // [3][N,256]
  float* keys  = ws + 75360000LL;       // [3][N,256]
  float* vals  = ws + 83040000LL;       // [3][N,256]
  float* query = ws + 90720000LL;       // [N,256]
  float* att   = ws + 93280000LL;       // [N,256]
  float* deg   = ws + 95840000LL;       // [N]
  float* bnst  = ws + 95850000LL;       // [512]

  float* z_t    = out;                  // [N,256]
  float* z_s    = out + 2560000LL;      // [N,256]
  float* o_mean = out + 5120000LL;      // [N,2000]
  float* o_disp = out + 25120000LL;
  float* o_pi   = out + 45120000LL;
  float* o_attn = out + 65120000LL;     // [N,3,8]

  const long NX = (long)NN * HV;
  const int gNX = (int)((NX + 255) / 256);
  const long ND = (long)NN * DDIM;
  const int gND = (int)((ND + 255) / 256);

  // ---- gcn_norm
  fill_k<<<(NN + 255) / 256, 256, 0, stream>>>(deg, 1.0f, NN);
  degcnt_k<<<(EE + 255) / 256, 256, 0, stream>>>(ei, deg);
  dinv_k<<<(NN + 255) / 256, 256, 0, stream>>>(deg);

  // ---- Teacher multi-scale SSGConv
  for (int s = 0; s < NSC; ++s) {
    int K = s + 1;
    float coef = (1.f - ALPHA_) / K;
    scale_k<<<gNX, 256, 0, stream>>>(bufH, x, ALPHA_, NX);
    const float* src = x;
    float* d0 = bufA; float* d1 = bufB;
    for (int t = 0; t < K; ++t) {
      selfloop_k<<<gNX, 256, 0, stream>>>(d0, src, deg);
      edgeprop_k<<<EE, 256, 0, stream>>>(ei, deg, src, d0);
      axpy_k<<<gNX, 256, 0, stream>>>(bufH, d0, coef, NX);
      src = d0; float* tmp = d0; d0 = d1; d1 = tmp;
    }
    int pb = P_SC0 + 10 * s;  // bn.b,bn.g,conv.b,conv.w,k.b,k.w,q.b,q.w,v.b,v.w
    float* fs = feats + (long)s * ND;
    launch_gemm<ACT_GELU>(stream, bufH, P(pb + 3), P(pb + 2), fs, NN, HV, DDIM);
    colstat_k<<<DDIM, 256, 0, stream>>>(fs, bnst);
    bnapply_k<<<gND, 256, 0, stream>>>(fs, bnst, P(pb + 1), P(pb + 0));
  }

  ln_k<0><<<NSC * NN, 256, 0, stream>>>(feats, P(P_T_LN_G), P(P_T_LN_B), nullptr, nf, DDIM);

  float* qtmp = bufA;
  fill_k<<<gND, 256, 0, stream>>>(query, 0.f, ND);
  for (int s = 0; s < NSC; ++s) {
    int pb = P_SC0 + 10 * s;
    launch_gemm<ACT_NONE>(stream, nf, P(pb + 7), P(pb + 6), qtmp, NN, DDIM, DDIM);
    axpy_k<<<gND, 256, 0, stream>>>(query, qtmp, 1.f / 3.f, ND);
    launch_gemm<ACT_NONE>(stream, nf + (long)s * ND, P(pb + 5), P(pb + 4),
                          keys + (long)s * ND, NN, DDIM, DDIM);
    launch_gemm<ACT_NONE>(stream, nf + (long)s * ND, P(pb + 9), P(pb + 8),
                          vals + (long)s * ND, NN, DDIM, DDIM);
  }
  attn_k<<<(NN * NHEADS + 255) / 256, 256, 0, stream>>>(query, keys, vals, feats, att, o_attn);
  launch_gemm<ACT_NONE>(stream, att, P(P_T_OUT_W), P(P_T_OUT_B), z_t, NN, DDIM, DDIM);

  // ---- Student INR
  float* enc = bufH;
  float* sh1 = bufH + 1280000LL;
  float* sh2 = bufH + 3840000LL;
  fourier_k<<<(NN * NFREQ + 255) / 256, 256, 0, stream>>>(pos, P(P_STU_B), enc);
  launch_gemm<ACT_SILU>(stream, enc, P(P_STU_H0_LIN_W), P(P_STU_H0_LIN_B), sh1, NN, ENCD, 256);
  ln_k<0><<<NN, 256, 0, stream>>>(sh1, P(P_STU_H0_LN_G), P(P_STU_H0_LN_B), nullptr, sh2, 256);
  launch_gemm<ACT_SILU>(stream, sh2, P(P_STU_H1_LIN_W), P(P_STU_H1_LIN_B), sh1, NN, 256, 256);
  ln_k<0><<<NN, 256, 0, stream>>>(sh1, P(P_STU_H1_LN_G), P(P_STU_H1_LN_B), nullptr, sh2, 256);
  launch_gemm<ACT_NONE>(stream, sh2, P(P_STU_OUT_W), P(P_STU_OUT_B), z_s, NN, 256, DDIM);

  // ---- ZINB decoder
  float* hcur = bufA;
  float* hnext = bufB;
  float* tA = bufH;
  float* tB = bufH + 10240000LL;
  launch_gemm<ACT_NONE>(stream, z_t, P(P_DEC_INP_LIN_W), P(P_DEC_INP_LIN_B), tA, NN, 256, 512);
  ln_k<1><<<NN, 256, 0, stream>>>(tA, P(P_DEC_INP_LN_G), P(P_DEC_INP_LN_B), nullptr, hcur, 512);

  auto resblock = [&](int base, int d) {
    launch_gemm<ACT_NONE>(stream, hcur, P(base + 1), P(base + 0), tA, NN, d, d);
    ln_k<1><<<NN, 256, 0, stream>>>(tA, P(base + 5), P(base + 4), nullptr, tB, d);
    launch_gemm<ACT_NONE>(stream, tB, P(base + 3), P(base + 2), tA, NN, d, d);
    ln_k<1><<<NN, 256, 0, stream>>>(tA, P(base + 7), P(base + 6), hcur, hnext, d);
    float* t = hcur; hcur = hnext; hnext = t;
  };
  resblock(P_RB0_LIN1_B, 512);
  launch_gemm<ACT_NONE>(stream, hcur, P(P_DEC_PROJ0_W), P(P_DEC_PROJ0_B), hnext, NN, 512, 1024);
  { float* t = hcur; hcur = hnext; hnext = t; }
  resblock(P_RB1_LIN1_B, 1024);
  launch_gemm<ACT_NONE>(stream, hcur, P(P_DEC_PROJ1_W), P(P_DEC_PROJ1_B), hnext, NN, 1024, 512);
  { float* t = hcur; hcur = hnext; hnext = t; }
  resblock(P_RB2_LIN1_B, 512);

  launch_gemm<ACT_SOFTPLUS>(stream, hcur, P(P_DEC_MEAN_W), P(P_DEC_MEAN_B), o_mean, NN, 512, HV);
  launch_gemm<ACT_SOFTPLUS>(stream, hcur, P(P_DEC_DISP_W), P(P_DEC_DISP_B), o_disp, NN, 512, HV);
  launch_gemm<ACT_SIGMOID >(stream, hcur, P(P_DEC_PI_W),   P(P_DEC_PI_B),   o_pi,   NN, 512, HV);
}
